// ConvDBN_71116068487527
// MI455X (gfx1250) — matrix operand order repaired and checked by reading, not compiled, not run on hardware
//
#include <hip/hip_runtime.h>
#include <hip/hip_bf16.h>
#include <cstdint>
#include <cstddef>

// ---------------------------------------------------------------------------
// CDNA5 (gfx1250) two-layer CRBM:
//   conv7x7(VALID) -> sigmoid((pre+b)/sigma^2) -> stochastic 3x3 Gumbel pool
// im2col + GEMM on v_wmma_f32_16x16x32_f16 (f16 in, f32 accumulate).
// B (weight) panels staged into LDS by the Tensor Data Mover, double buffered;
// A streamed from global as b128 loads. Hot loop: ds_load(B) + 4x b128(A) +
// 2x WMMA per k-chunk.
// ---------------------------------------------------------------------------

typedef __attribute__((ext_vector_type(16))) _Float16     v16h;
typedef __attribute__((ext_vector_type(8)))  _Float16     v8h;
typedef __attribute__((ext_vector_type(8)))  float        v8f;
typedef __attribute__((ext_vector_type(4)))  unsigned int v4u;
typedef __attribute__((ext_vector_type(8)))  int          v8i;
typedef __attribute__((ext_vector_type(4)))  int          v4i;

#if defined(__has_builtin)
#if __has_builtin(__builtin_amdgcn_tensor_load_to_lds) && \
    __has_builtin(__builtin_amdgcn_s_wait_tensorcnt)
#define HAVE_TDM 1
#else
#define HAVE_TDM 0
#endif
#else
#define HAVE_TDM 0
#endif

// Problem constants
static constexpr int NB = 64;                   // batch
// Layer 1
static constexpr int C0 = 3,  HI1 = 96, WI1 = 96;
static constexpr int C1 = 96, HO1 = 90, WO1 = 90;
static constexpr int K1 = 147, KC1 = 5, KP1 = 160;   // K padded to 5 chunks of 32
static constexpr int HB1 = 30, WB1 = 30;
// Layer 2
static constexpr int C2 = 192, HO2 = 24, WO2 = 24;
static constexpr int K2 = 4704, KC2 = 147, KP2 = 4704; // 96*49 = 147*32 exactly
static constexpr int HB2 = 8,  WB2 = 8;

// ---------------------------------------------------------------------------
// TDM: 1-D contiguous copy global -> LDS.  D# group0: count=1, lds_addr,
// global_addr, type=2.  Group1: data_size=3 (8B), tensor_dim0 = tile_dim0 =
// stride = bytes/8, tensor_dim1 = tile_dim1 = 1.  Groups 2/3 unused (zeros).
// This toolchain exposes the 6-arg builtin:
//   (uint32x4 g0, int32x8 g1, int32x4, int32x4, int32x8, i32 cpol)
// ---------------------------------------------------------------------------
__device__ __forceinline__ void tdm_load_1d(void* lds_ptr, const void* gptr,
                                            uint32_t bytes) {
#if HAVE_TDM
    const uint32_t units = bytes >> 3;                      // 8-byte elements
    const uint64_t ga = (uint64_t)(uintptr_t)gptr;
    const uint32_t la = (uint32_t)(uintptr_t)lds_ptr;       // LDS byte offset
    v4u g0;
    g0[0] = 1u;                                             // count=1 (valid)
    g0[1] = la;                                             // lds_addr
    g0[2] = (uint32_t)ga;                                   // global_addr lo
    g0[3] = (uint32_t)((ga >> 32) & 0x01FFFFFFu) | (2u << 30); // addr hi | type=2
    v8i g1;
    g1[0] = (int)(3u << 16);                                // data_size=3 (8B)
    g1[1] = (int)((units & 0xFFFFu) << 16);                 // tensor_dim0 lo16
    g1[2] = (int)(((units >> 16) & 0xFFFFu) | (1u << 16));  // td0 hi | td1=1 lo
    g1[3] = (int)((units & 0xFFFFu) << 16);                 // tile_dim0
    g1[4] = 1;                                              // tile_dim1 = 1
    g1[5] = (int)units;                                     // td0_stride lo32
    g1[6] = 0;
    g1[7] = 0;
    v4i z4 = {0, 0, 0, 0};
    v8i z8 = {0, 0, 0, 0, 0, 0, 0, 0};
    __builtin_amdgcn_tensor_load_to_lds(g0, g1, z4, z4, z8, 0);
#else
    (void)lds_ptr; (void)gptr; (void)bytes;
#endif
}

// ---------------------------------------------------------------------------
// f32 -> f16 elementwise convert
// ---------------------------------------------------------------------------
__global__ void cvt_f32_to_f16(const float* __restrict__ in,
                               _Float16* __restrict__ out, int n) {
    int i = blockIdx.x * blockDim.x + threadIdx.x;
    if (i < n) out[i] = (_Float16)in[i];
}

// ---------------------------------------------------------------------------
// im2col: A[m*KPAD + K] = x[b, c, oy+dy, ox+dx]  (f16), zero-pad K >= KMAX.
// ---------------------------------------------------------------------------
template<int CIN, int HIN, int WIN, int HOUT, int WOUT, int KPAD, int KMAX>
__global__ void im2col_f16(const _Float16* __restrict__ xin,
                           _Float16* __restrict__ A, size_t total) {
    size_t e = (size_t)blockIdx.x * blockDim.x + threadIdx.x;
    if (e >= total) return;
    const int    K = (int)(e % KPAD);
    const size_t m = e / KPAD;
    _Float16 v = (_Float16)0.0f;
    if (K < KMAX) {
        const int b  = (int)(m / (HOUT * WOUT));
        const int rr = (int)(m - (size_t)b * (HOUT * WOUT));
        const int oy = rr / WOUT;
        const int ox = rr - oy * WOUT;
        const int c  = K / 49;
        const int r  = K - c * 49;
        const int dy = r / 7;
        const int dx = r - dy * 7;
        v = xin[(((size_t)b * CIN + c) * HIN + (oy + dy)) * WIN + (ox + dx)];
    }
    A[e] = v;
}

// ---------------------------------------------------------------------------
// Pack weights [Cout, Cin*49] (OIHW flat) into WMMA B-fragment order:
//   wpack[((tileN*KCH + kc)*32 + lane)*16 + j] = W[n, K],
//   n = tileN*16 + (lane&15), K = kc*32 + (lane>>4)*16 + j  (0 if K>=KMAX).
// ---------------------------------------------------------------------------
template<int KCH, int KMAX>
__global__ void pack_weights(const float* __restrict__ Wsrc,
                             _Float16* __restrict__ wpack, int ntiles) {
    int e = blockIdx.x * blockDim.x + threadIdx.x;
    int total = ntiles * KCH * 512;
    if (e >= total) return;
    int j    = e & 15;
    int lane = (e >> 4) & 31;
    int rest = e >> 9;
    int kc   = rest % KCH;
    int tn   = rest / KCH;
    int n    = tn * 16 + (lane & 15);
    int K    = kc * 32 + (lane >> 4) * 16 + j;
    float v  = (K < KMAX) ? Wsrc[(size_t)n * KMAX + K] : 0.0f;
    wpack[e] = (_Float16)v;
}

// ---------------------------------------------------------------------------
// GEMM + bias + /sigma^2 + sigmoid -> f16 prob map (NCHW).
//   C[M,N] = A[M,K] * W[K,N],  M = NB*HOUT*WOUT, N = COUT, K = KGRP*GRPS*32.
// 8 waves/block, each wave computes TWO 16x16 M-tiles; all waves share the
// tileN weight panel, staged into LDS in KGRP-chunk slices by the TDM
// (double buffered, s_wait_tensorcnt + block barriers for hand-off).
// ---------------------------------------------------------------------------
template<int HOUT, int WOUT, int KGRP, int GRPS, int KPAD, int COUT>
__global__ __launch_bounds__(256)
void gemm_sigmoid_wmma(const _Float16* __restrict__ A,
                       const _Float16* __restrict__ wpack,
                       const float* __restrict__ bias,
                       _Float16* __restrict__ prob) {
    constexpr int KCH       = KGRP * GRPS;       // total k-chunks
    constexpr int SLICE_ELT = KGRP * 512;        // f16 per slice
    __shared__ __align__(32) _Float16 bsm[2][SLICE_ELT];

    const int lane   = threadIdx.x & 31;
    const int wave   = threadIdx.x >> 5;
    const int tileM0 = (blockIdx.x * 8 + wave) * 2;   // two M-tiles per wave
    const int tileN  = blockIdx.y;

    // Per-lane A row pointers; lane-half K base: lane<16 -> K 0..15,
    // lane>=16 -> 16..31 (contiguous, matches packed B; row = lane&15).
    const int khalf = (lane >> 4) * 16;
    const _Float16* a0 =
        A + ((size_t)(tileM0 * 16 + (lane & 15))) * KPAD + khalf;
    const _Float16* a1 = a0 + (size_t)16 * KPAD;
    const _Float16* wt = wpack + (size_t)tileN * KCH * 512;   // panel base

#if HAVE_TDM
    if (wave == 0)                                    // preload slice 0
        tdm_load_1d(&bsm[0][0], wt, SLICE_ELT * 2);
#endif

    v8f acc0 = {}, acc1 = {};
    for (int g = 0; g < GRPS; ++g) {
        __syncthreads();          // buffer (g&1) free from iteration g-2
#if HAVE_TDM
        if (wave == 0) {
            if (g + 1 < GRPS) {   // overlap next slice DMA with this compute
                tdm_load_1d(&bsm[(g + 1) & 1][0],
                            wt + (size_t)(g + 1) * SLICE_ELT, SLICE_ELT * 2);
                __builtin_amdgcn_s_wait_tensorcnt(1);  // slice g landed
            } else {
                __builtin_amdgcn_s_wait_tensorcnt(0);
            }
        }
#else
        {   // cooperative fallback copy (256 threads, 16B each)
            const _Float16* src = wt + (size_t)g * SLICE_ELT;
            for (int e = threadIdx.x; e < SLICE_ELT / 8; e += 256)
                *(v8h*)&bsm[g & 1][(size_t)e * 8] =
                    *(const v8h*)&src[(size_t)e * 8];
        }
#endif
        __syncthreads();          // slice g visible to all waves

        const _Float16* bs = &bsm[g & 1][(size_t)lane * 16];
#pragma unroll
        for (int kk = 0; kk < KGRP; ++kk) {
            const v16h bfrag = *(const v16h*)(bs + (size_t)kk * 512);
            const int  kc    = g * KGRP + kk;
            const v8h p0 = *(const v8h*)(a0 + (size_t)kc * 32);
            const v8h p1 = *(const v8h*)(a0 + (size_t)kc * 32 + 8);
            const v8h q0 = *(const v8h*)(a1 + (size_t)kc * 32);
            const v8h q1 = *(const v8h*)(a1 + (size_t)kc * 32 + 8);
            v16h af0, af1;
#pragma unroll
            for (int j = 0; j < 8; ++j) {
                af0[j] = p0[j]; af0[j + 8] = p1[j];
                af1[j] = q0[j]; af1[j + 8] = q1[j];
            }
            acc0 = __builtin_amdgcn_wmma_f32_16x16x32_f16(
                false, af0, false, bfrag, (short)0, acc0, false, false);
            acc1 = __builtin_amdgcn_wmma_f32_16x16x32_f16(
                false, af1, false, bfrag, (short)0, acc1, false, false);
        }
    }

    // Epilogue. C/D layout: N = lane&15, M = r + (lane>=16 ? 8 : 0)
    const int n    = tileN * 16 + (lane & 15);
    const float bn = bias[n];
#pragma unroll
    for (int t = 0; t < 2; ++t) {
        const v8f& acc = t ? acc1 : acc0;
        const int mbase = (tileM0 + t) * 16 + ((lane >> 4) ? 8 : 0);
#pragma unroll
        for (int r2 = 0; r2 < 8; ++r2) {
            const int m  = mbase + r2;
            const int bb = m / (HOUT * WOUT);
            const int rr = m - bb * (HOUT * WOUT);
            const int yy = rr / WOUT;
            const int xx = rr - yy * WOUT;
            const float pre = (acc[r2] + bn) * 25.0f;        // 1 / sigma^2
            const float p   = 1.0f / (1.0f + __expf(-pre));  // sigmoid
            prob[(((size_t)bb * COUT + n) * HOUT + yy) * WOUT + xx] = (_Float16)p;
        }
    }
}

// ---------------------------------------------------------------------------
// Stochastic multinomial 3x3 pool: argmax of log(p+eps)+Gumbel, output winner
// prob. Deterministic counter-based hash RNG (same inputs -> same output).
// ---------------------------------------------------------------------------
__device__ __forceinline__ uint32_t hash32(uint32_t x) {
    x ^= x >> 16; x *= 0x7feb352du;
    x ^= x >> 15; x *= 0x846ca68bu;
    x ^= x >> 16;
    return x;
}

template<int C, int HB, int WB, int HIN, int WIN, typename OutT>
__global__ void gumbel_pool(const _Float16* __restrict__ prob,
                            OutT* __restrict__ out, uint32_t seed) {
    const int idx   = blockIdx.x * blockDim.x + threadIdx.x;
    const int total = NB * C * HB * WB;
    if (idx >= total) return;
    const int xb = idx % WB;
    int t = idx / WB;
    const int yb = t % HB;  t /= HB;
    const int c  = t % C;
    const int bb = t / C;

    const _Float16* base =
        prob + (((size_t)bb * C + c) * HIN + yb * 3) * WIN + xb * 3;

    float best = -3.0e38f, bp = 0.0f;
#pragma unroll
    for (int s = 0; s < 9; ++s) {
        const int dy = s / 3, dx = s - dy * 3;
        const float p = (float)base[(size_t)dy * WIN + dx];
        const uint32_t h = hash32((uint32_t)idx * 9u + (uint32_t)s + seed);
        const float u = (float)(h >> 8) * (1.0f / 16777216.0f);
        const float g = -__logf(-__logf(u + 1e-20f) + 1e-20f);   // Gumbel(0,1)
        const float sc = __logf(p + 1e-8f) + g;
        if (sc > best) { best = sc; bp = p; }
    }
    out[idx] = (OutT)bp;
}

// ---------------------------------------------------------------------------
// Host launcher
// ---------------------------------------------------------------------------
extern "C" void kernel_launch(void* const* d_in, const int* in_sizes, int n_in,
                              void* d_out, int out_size, void* d_ws, size_t ws_size,
                              hipStream_t stream) {
    const float* x  = (const float*)d_in[0];   // [64,3,96,96]
    const float* W1 = (const float*)d_in[1];   // [96,3,7,7]
    const float* b1 = (const float*)d_in[2];   // [96]
    const float* W2 = (const float*)d_in[3];   // [192,96,7,7]
    const float* b2 = (const float*)d_in[4];   // [192]

    // Workspace carve-up (256B aligned).  A-region and prob-region are reused
    // across layers (stream order serializes producer/consumer).
    char* ws = (char*)d_ws;
    size_t off = 0;
    auto carve = [&](size_t bytes) {
        size_t o = off;
        off = (off + bytes + 255) & ~(size_t)255;
        return o;
    };
    const size_t M1 = (size_t)NB * HO1 * WO1;      // 518,400
    const size_t M2 = (size_t)NB * HO2 * WO2;      // 36,864
    const size_t nX   = (size_t)NB * C0 * HI1 * WI1;
    const size_t nW1p = (size_t)(C1 / 16) * KC1 * 512;
    const size_t nW2p = (size_t)(C2 / 16) * KC2 * 512;
    const size_t nA1  = M1 * KP1;                  //  82,944,000
    const size_t nA2  = M2 * KP2;                  // 173,408,256
    const size_t nA   = (nA1 > nA2) ? nA1 : nA2;
    const size_t nP1  = (size_t)NB * C1 * HO1 * WO1;
    const size_t nP2  = (size_t)NB * C2 * HO2 * WO2;
    const size_t nP   = (nP1 > nP2) ? nP1 : nP2;
    const size_t nH1  = (size_t)NB * C1 * HB1 * WB1;

    _Float16* xf16 = (_Float16*)(ws + carve(nX   * 2));
    _Float16* w1p  = (_Float16*)(ws + carve(nW1p * 2));
    _Float16* w2p  = (_Float16*)(ws + carve(nW2p * 2));
    _Float16* h1   = (_Float16*)(ws + carve(nH1  * 2));
    _Float16* Abuf = (_Float16*)(ws + carve(nA   * 2));   // A1 then A2
    _Float16* Pbuf = (_Float16*)(ws + carve(nP   * 2));   // prob1 then prob2

    // 1) x -> f16; pack weights into B-fragment order
    cvt_f32_to_f16<<<(int)((nX + 255) / 256), 256, 0, stream>>>(x, xf16, (int)nX);
    pack_weights<KC1, K1><<<(int)((nW1p + 255) / 256), 256, 0, stream>>>(W1, w1p, C1 / 16);
    pack_weights<KC2, K2><<<(int)((nW2p + 255) / 256), 256, 0, stream>>>(W2, w2p, C2 / 16);

    // 2) layer 1: im2col -> GEMM(+sigmoid) -> stochastic pool
    im2col_f16<C0, HI1, WI1, HO1, WO1, KP1, K1>
        <<<(int)((nA1 + 255) / 256), 256, 0, stream>>>(xf16, Abuf, nA1);
    {
        dim3 grid((int)(M1 / 16 / 16), C1 / 16);   // (2025, 6): 16 M-tiles/block
        gemm_sigmoid_wmma<HO1, WO1, /*KGRP=*/KC1, /*GRPS=*/1, KP1, C1>
            <<<grid, 256, 0, stream>>>(Abuf, w1p, b1, Pbuf);
    }
    gumbel_pool<C1, HB1, WB1, HO1, WO1, _Float16>
        <<<(int)((nH1 + 255) / 256), 256, 0, stream>>>(Pbuf, h1, 0x9E3779B9u);

    // 3) layer 2: im2col -> GEMM(+sigmoid) -> stochastic pool -> d_out
    im2col_f16<C1, HB1, WB1, HO2, WO2, KP2, K2>
        <<<(int)((nA2 + 255) / 256), 256, 0, stream>>>(h1, Abuf, nA2);
    {
        dim3 grid((int)(M2 / 16 / 16), C2 / 16);   // (144, 12)
        gemm_sigmoid_wmma<HO2, WO2, /*KGRP=*/7, /*GRPS=*/21, KP2, C2>
            <<<grid, 256, 0, stream>>>(Abuf, w2p, b2, Pbuf);
    }
    {
        const size_t tot = (size_t)NB * C2 * HB2 * WB2;    // 786,432 == out_size
        gumbel_pool<C2, HB2, WB2, HO2, WO2, float>
            <<<(int)((tot + 255) / 256), 256, 0, stream>>>(Pbuf, (float*)d_out,
                                                           0x85EBCA6Bu);
    }
}
